// MultiHeadSelfAttentionBlock_60722247631582
// MI455X (gfx1250) — matrix operand order repaired and checked by reading, not compiled, not run on hardware
//
#include <hip/hip_runtime.h>
#include <hip/hip_bf16.h>
#include <math.h>

// ---------------------------------------------------------------------------
// CDNA5 (gfx1250) wave32 WMMA attention block.
// Matmuls: v_wmma_f32_16x16x32_bf16. Tiles staged in LDS via async
// global->LDS (ASYNCcnt) when available; fragments read as b128.
// ---------------------------------------------------------------------------

typedef __attribute__((ext_vector_type(16))) __bf16 v16bf;
typedef __attribute__((ext_vector_type(8)))  float  v8f;
typedef int v4i __attribute__((vector_size(16)));

#define BB 8
#define NN 2048
#define DIN 256
#define DV 256
#define HH 8
#define HD 32
#define DFF 512
#define ROWS (BB * NN)   // 16384

#if __has_builtin(__builtin_amdgcn_global_load_async_to_lds_b128)
#define ASYNC_LDS 1
#endif

// Copy 16 bytes global -> LDS (async if available).
__device__ __forceinline__ void cp16(void* lds, const void* g) {
#ifdef ASYNC_LDS
  __builtin_amdgcn_global_load_async_to_lds_b128(
      (__attribute__((address_space(1))) v4i*)(void*)g,
      (__attribute__((address_space(3))) v4i*)lds, 0, 0);
#else
  *(uint4*)lds = *(const uint4*)g;
#endif
}

// Producer side: make this wave's copies visible, then block barrier.
__device__ __forceinline__ void async_join() {
#ifdef ASYNC_LDS
#if __has_builtin(__builtin_amdgcn_s_wait_asynccnt)
  __builtin_amdgcn_s_wait_asynccnt(0);
#else
  asm volatile("s_wait_asynccnt 0x0" ::: "memory");
#endif
#endif
  __syncthreads();
}

// Consumer side: drain this wave's LDS reads, then block barrier.
__device__ __forceinline__ void lds_release() {
  asm volatile("s_wait_dscnt 0x0" ::: "memory");
  __syncthreads();
}

__device__ __forceinline__ v8f wmma_bf16(v16bf a, v16bf b, v8f c) {
  return __builtin_amdgcn_wmma_f32_16x16x32_bf16(
      /*neg_a=*/false, a, /*neg_b=*/false, b,
      /*c_mod=*/(short)0, c, /*reuse_a=*/false, /*reuse_b=*/false);
}

// A-fragment: rows m0..m0+15, contraction cols k0..k0+31 of row-major A[ldk].
// ISA 16-bit A layout: lanes 0-15 hold K {0..7,16..23}, lanes 16-31 {8..15,24..31}.
// Two 16B loads per lane.
__device__ __forceinline__ v16bf load_a(const __bf16* __restrict__ A, int ldk,
                                        int m0, int k0) {
  int l = threadIdx.x & 31;
  const __bf16* p = A + (size_t)(m0 + (l & 15)) * ldk + k0 + ((l >> 4) << 3);
  v16bf a;
  *(uint4*)&a       = *(const uint4*)p;
  *((uint4*)&a + 1) = *(const uint4*)(p + 16);
  return a;
}

// B-fragment from k-contiguous storage Bt[n][k] (global or LDS tile):
// contraction rows k0..k0+31, cols n0..n0+15. Two 16B loads per lane.
// ISA B layout: lanes 0-15 hold K {0..7,16..23}, lanes 16-31 {8..15,24..31}.
__device__ __forceinline__ v16bf frag_bt(const __bf16* Bt, int ldk, int k0,
                                         int n0) {
  int l = threadIdx.x & 31;
  const __bf16* p = Bt + (size_t)(n0 + (l & 15)) * ldk + k0 + ((l >> 4) << 3);
  v16bf b;
  *(uint4*)&b       = *(const uint4*)p;
  *((uint4*)&b + 1) = *(const uint4*)(p + 16);
  return b;
}

// ---------------------------------------------------------------------------
// LayerNorm(256) + cast to bf16. One 256-thread block per row.
// ---------------------------------------------------------------------------
__global__ __launch_bounds__(256) void k_ln_cast(const float* __restrict__ X,
                                                 const float* __restrict__ g,
                                                 const float* __restrict__ bt,
                                                 __bf16* __restrict__ Xn) {
  __shared__ float sm[8], sv[8];
  int row = blockIdx.x, t = threadIdx.x;
  float x = X[(size_t)row * 256 + t];
  float s = x, s2 = x * x;
#pragma unroll
  for (int off = 16; off > 0; off >>= 1) {
    s += __shfl_xor(s, off, 32);
    s2 += __shfl_xor(s2, off, 32);
  }
  if ((t & 31) == 0) { sm[t >> 5] = s; sv[t >> 5] = s2; }
  __syncthreads();
  float S = 0.f, S2 = 0.f;
#pragma unroll
  for (int i = 0; i < 8; ++i) { S += sm[i]; S2 += sv[i]; }
  float mu = S * (1.f / 256.f);
  float var = S2 * (1.f / 256.f) - mu * mu;
  float inv = rsqrtf(var + 1e-5f);
  Xn[(size_t)row * 256 + t] = (__bf16)((x - mu) * inv * g[t] + bt[t]);
}

// f32 -> bf16 cast
__global__ void k_cast(const float* __restrict__ src, __bf16* __restrict__ dst,
                       int n) {
  int i = blockIdx.x * 256 + threadIdx.x;
  if (i < n) dst[i] = (__bf16)src[i];
}

// f32 W[K][N] -> bf16 Wt[N][K] (transpose so B-fragments are k-contiguous)
__global__ void k_castT(const float* __restrict__ W, __bf16* __restrict__ Wt,
                        int K, int N) {
  int i = blockIdx.x * 256 + threadIdx.x;
  if (i < K * N) {
    int k = i / N, n = i - k * N;
    Wt[(size_t)n * K + k] = (__bf16)W[i];
  }
}

// ---------------------------------------------------------------------------
// Generic bf16 WMMA GEMM: C[M,N] = epi(A[M,K] @ W[K,N] + bias), W given
// transposed (Wt[N][K]). Block 256 = 8 waves; wave -> 16x64, block -> 128x64.
// B tile (64n x 32k, 4KB) async-staged in LDS, shared by all 8 waves.
// mode 0: f32 out  | mode 1: exact-GeLU -> bf16 out | mode 2: +res -> f32 out
// ---------------------------------------------------------------------------
__global__ __launch_bounds__(256) void k_gemm(
    const __bf16* __restrict__ A, const __bf16* __restrict__ Wt,
    const float* __restrict__ bias, const float* __restrict__ res,
    float* __restrict__ outF, __bf16* __restrict__ outB,
    int M, int N, int K, int mode) {
  __shared__ __bf16 btile[64][32];
  int t = threadIdx.x, w = t >> 5, l = t & 31;
  int m0 = blockIdx.y * 128 + w * 16;
  int n0 = blockIdx.x * 64;
  v8f acc[4] = {};
  for (int k0 = 0; k0 < K; k0 += 32) {
    // 256 threads x 16B = 4KB tile: rows n0..n0+63, cols k0..k0+31 of Wt
    cp16(&btile[t >> 2][(t & 3) * 8],
         Wt + (size_t)(n0 + (t >> 2)) * K + k0 + (t & 3) * 8);
    async_join();
    v16bf a = load_a(A, K, m0, k0);
#pragma unroll
    for (int j = 0; j < 4; ++j) {
      v16bf b = frag_bt(&btile[0][0], 32, 0, j * 16);
      acc[j] = wmma_bf16(a, b, acc[j]);
    }
    lds_release();
  }
  int n = l & 15, mh = (l >> 4) << 3;
#pragma unroll
  for (int j = 0; j < 4; ++j) {
    int nc = n0 + j * 16 + n;
    float bb = bias[nc];
#pragma unroll
    for (int r = 0; r < 8; ++r) {
      size_t idx = (size_t)(m0 + mh + r) * N + nc;
      float v = acc[j][r] + bb;
      if (mode == 0) outF[idx] = v;
      else if (mode == 1)
        outB[idx] = (__bf16)(0.5f * v * (1.f + erff(v * 0.70710678f)));
      else outF[idx] = v + res[idx];
    }
  }
}

// ---------------------------------------------------------------------------
// Pack per-head bf16 tensors: Q,K -> [B,H,N,32]; V -> transposed [B,H,32,N].
// ---------------------------------------------------------------------------
__global__ void k_pack(const float* __restrict__ Qf, const float* __restrict__ Kf,
                       const float* __restrict__ Vf, __bf16* __restrict__ Qbf,
                       __bf16* __restrict__ Kbf, __bf16* __restrict__ Vt) {
  int row = blockIdx.x, c = threadIdx.x;     // row in [0,16384), c in [0,256)
  int b = row >> 11, n = row & (NN - 1);
  int h = c >> 5, d = c & 31;
  size_t src = (size_t)row * 256 + c;
  size_t bh = (size_t)(b * HH + h);
  Qbf[(bh * NN + n) * HD + d] = (__bf16)Qf[src];
  Kbf[(bh * NN + n) * HD + d] = (__bf16)Kf[src];
  Vt[(bh * HD + d) * NN + n]  = (__bf16)Vf[src];
}

// ---------------------------------------------------------------------------
// Flash attention. Block = 4 waves sharing one (b,h); wave w owns query tile
// qt = 4*blk + w (16 queries). Per 32-key step: K tile (2KB, contiguous) and
// V tile (32 hd rows x 64B) async-staged in LDS; 2 WMMA for S; online softmax
// via shfl in C-layout lane halves; P transposed C->A layout through LDS;
// 2 WMMA for A@V. Epilogue adds per-head residual Q and 1/rowsum.
// ---------------------------------------------------------------------------
__global__ __launch_bounds__(128) void k_attn(
    const __bf16* __restrict__ Qbf, const __bf16* __restrict__ Kbf,
    const __bf16* __restrict__ Vt, const float* __restrict__ Qf,
    float* __restrict__ O) {
  __shared__ __bf16 kt[32][32];          // [key][hd]   2KB
  __shared__ __bf16 vt[32][32];          // [hd][key]   2KB
  __shared__ float sp[4][16][33];        // per-wave P transpose staging
  int t = threadIdx.x, wave = t >> 5, l = t & 31;
  int bh = blockIdx.x >> 5;              // 32 blocks per (b,h)
  int qt = (blockIdx.x & 31) * 4 + wave; // query tile in [0,128)
  int b = bh >> 3, h = bh & 7;
  const __bf16* Qh = Qbf + (size_t)bh * NN * HD;
  const __bf16* Kh = Kbf + (size_t)bh * NN * HD;
  const __bf16* Vh = Vt + (size_t)bh * HD * NN;

  v16bf aq = load_a(Qh, HD, qt * 16, 0); // 16x32 Q fragment (whole head dim)
  v8f o0 = {}, o1 = {};
  float rmax[8], rsum[8];
#pragma unroll
  for (int r = 0; r < 8; ++r) { rmax[r] = -3.0e38f; rsum[r] = 0.f; }
  float (*spw)[33] = sp[wave];

  for (int kb = 0; kb < NN; kb += 32) {
    // K tile: rows kb..kb+31 of Kh = one contiguous 2KB chunk.
    cp16(((__bf16*)kt) + t * 8, Kh + (size_t)kb * HD + t * 8);
    // V tile: vt[hd][0..31] <- Vh[hd][kb..kb+31]; 4 threads x 16B per row.
    cp16(&vt[t >> 2][(t & 3) * 8],
         Vh + (size_t)(t >> 2) * NN + kb + (t & 3) * 8);
    async_join();

    v16bf bk0 = frag_bt(&kt[0][0], HD, 0, 0);   // keys 0..15 (local)
    v16bf bk1 = frag_bt(&kt[0][0], HD, 0, 16);  // keys 16..31
    v8f z = {};
    v8f s0 = wmma_bf16(aq, bk0, z);
    v8f s1 = wmma_bf16(aq, bk1, z);
#pragma unroll
    for (int r = 0; r < 8; ++r) {        // online softmax per output row
      float sa = s0[r] * 0.0625f;        // 1/sqrt(D_V) = 1/16
      float sb = s1[r] * 0.0625f;
      float mx = fmaxf(sa, sb);
#pragma unroll
      for (int off = 1; off < 16; off <<= 1)
        mx = fmaxf(mx, __shfl_xor(mx, off, 32));
      float nm = fmaxf(rmax[r], mx);
      float alpha = __expf(rmax[r] - nm);
      rmax[r] = nm;
      float p0 = __expf(sa - nm), p1 = __expf(sb - nm);
      float ts = p0 + p1;
#pragma unroll
      for (int off = 1; off < 16; off <<= 1) ts += __shfl_xor(ts, off, 32);
      rsum[r] = rsum[r] * alpha + ts;
      o0[r] *= alpha;
      o1[r] *= alpha;
      int m = r + ((l >> 4) << 3);       // C-layout row for this VGPR
      spw[m][l & 15] = p0;               // stage P tile 16x32 in LDS
      spw[m][16 + (l & 15)] = p1;
    }
    asm volatile("s_wait_dscnt 0x0" ::: "memory");  // cross-lane visibility
    v16bf ap;                            // re-read P in A-fragment layout
    {
      int m = l & 15, kh = (l >> 4) << 3;
#pragma unroll
      for (int i = 0; i < 8; ++i) {
        ap[i]     = (__bf16)spw[m][kh + i];
        ap[8 + i] = (__bf16)spw[m][kh + 16 + i];
      }
    }
    v16bf bv0 = frag_bt(&vt[0][0], 32, 0, 0);   // 32 keys x hd 0..15
    v16bf bv1 = frag_bt(&vt[0][0], 32, 0, 16);  // hd 16..31
    lds_release();                        // reads drained; tiles reusable
    o0 = wmma_bf16(ap, bv0, o0);
    o1 = wmma_bf16(ap, bv1, o1);
  }

  int n = l & 15, mh = (l >> 4) << 3;
#pragma unroll
  for (int r = 0; r < 8; ++r) {
    int m = qt * 16 + mh + r;
    size_t base = ((size_t)(b * NN + m)) * DV + h * HD;
    float inv = 1.f / rsum[r];
    O[base + n]      = Qf[base + n]      + o0[r] * inv;   // per-head residual
    O[base + 16 + n] = Qf[base + 16 + n] + o1[r] * inv;
  }
}

// ---------------------------------------------------------------------------
// Host-side launch
// ---------------------------------------------------------------------------
extern "C" void kernel_launch(void* const* d_in, const int* in_sizes, int n_in,
                              void* d_out, int out_size, void* d_ws, size_t ws_size,
                              hipStream_t stream) {
  const float* x    = (const float*)d_in[0];
  const float* y    = (const float*)d_in[1];
  const float* Wq   = (const float*)d_in[2];
  const float* bq   = (const float*)d_in[3];
  const float* Wk   = (const float*)d_in[4];
  const float* bk   = (const float*)d_in[5];
  const float* Wv   = (const float*)d_in[6];
  const float* bv   = (const float*)d_in[7];
  const float* W1   = (const float*)d_in[8];
  const float* b1   = (const float*)d_in[9];
  const float* W2   = (const float*)d_in[10];
  const float* b2   = (const float*)d_in[11];
  const float* ln0g = (const float*)d_in[12];
  const float* ln0b = (const float*)d_in[13];
  const float* ln1g = (const float*)d_in[14];
  const float* ln1b = (const float*)d_in[15];

  char* ws = (char*)d_ws;
  const size_t MB = 1u << 20;
  __bf16* ybf  = (__bf16*)(ws + 0 * MB);      // 8 MiB
  __bf16* xnbf = (__bf16*)(ws + 8 * MB);      // 8 MiB (reused as On after attn)
  __bf16* wqb  = (__bf16*)(ws + 16 * MB);     // 128 KiB each (transposed)
  __bf16* wkb  = (__bf16*)(ws + 16 * MB + 128 * 1024);
  __bf16* wvb  = (__bf16*)(ws + 16 * MB + 256 * 1024);
  __bf16* w1b  = (__bf16*)(ws + 16 * MB + 384 * 1024);  // 256 KiB
  __bf16* w2b  = (__bf16*)(ws + 16 * MB + 640 * 1024);  // 256 KiB
  float*  Qf   = (float*)(ws + 17 * MB);      // 16 MiB
  float*  Kf   = (float*)(ws + 33 * MB);      // 16 MiB (reused as O)
  float*  Vf   = (float*)(ws + 49 * MB);      // 16 MiB (reused as ff1)
  __bf16* Qbf  = (__bf16*)(ws + 65 * MB);     // 8 MiB
  __bf16* Kbf  = (__bf16*)(ws + 73 * MB);     // 8 MiB
  __bf16* Vt   = (__bf16*)(ws + 81 * MB);     // 8 MiB ; total 89 MiB
  float*  O    = Kf;                           // alias (Kf dead after k_pack)
  __bf16* ff1  = (__bf16*)Vf;                  // alias (Vf dead after k_pack)
  __bf16* Onbf = xnbf;                         // alias (xn dead after Q gemm)

  // 1) LN0(x) -> bf16 ; casts (weights transposed to [N][K])
  k_ln_cast<<<ROWS, 256, 0, stream>>>(x, ln0g, ln0b, xnbf);
  k_cast<<<(ROWS * DIN + 255) / 256, 256, 0, stream>>>(y, ybf, ROWS * DIN);
  k_castT<<<(DIN * DV + 255) / 256, 256, 0, stream>>>(Wq, wqb, DIN, DV);
  k_castT<<<(DIN * DV + 255) / 256, 256, 0, stream>>>(Wk, wkb, DIN, DV);
  k_castT<<<(DIN * DV + 255) / 256, 256, 0, stream>>>(Wv, wvb, DIN, DV);
  k_castT<<<(DV * DFF + 255) / 256, 256, 0, stream>>>(W1, w1b, DV, DFF);
  k_castT<<<(DFF * DV + 255) / 256, 256, 0, stream>>>(W2, w2b, DFF, DV);

  // 2) Q/K/V projections (WMMA GEMMs)
  dim3 gp(DV / 64, ROWS / 128);
  k_gemm<<<gp, 256, 0, stream>>>(xnbf, wqb, bq, nullptr, Qf, nullptr,
                                 ROWS, DV, DIN, 0);
  k_gemm<<<gp, 256, 0, stream>>>(ybf, wkb, bk, nullptr, Kf, nullptr,
                                 ROWS, DV, DIN, 0);
  k_gemm<<<gp, 256, 0, stream>>>(ybf, wvb, bv, nullptr, Vf, nullptr,
                                 ROWS, DV, DIN, 0);

  // 3) Per-head bf16 pack (V transposed for contiguous B-fragments)
  k_pack<<<ROWS, 256, 0, stream>>>(Qf, Kf, Vf, Qbf, Kbf, Vt);

  // 4) Flash attention + per-head residual -> O (f32)
  k_attn<<<BB * HH * 32, 128, 0, stream>>>(Qbf, Kbf, Vt, Qf, O);

  // 5) LN1(O) -> bf16
  k_ln_cast<<<ROWS, 256, 0, stream>>>(O, ln1g, ln1b, Onbf);

  // 6) FFN: gelu(On@W1+b1) -> bf16 ; (ff1@W2+b2)+O -> d_out (f32)
  dim3 g1(DFF / 64, ROWS / 128);
  k_gemm<<<g1, 256, 0, stream>>>(Onbf, w1b, b1, nullptr, nullptr, ff1,
                                 ROWS, DFF, DV, 1);
  dim3 g2(DV / 64, ROWS / 128);
  k_gemm<<<g2, 256, 0, stream>>>(ff1, w2b, b2, O, (float*)d_out, nullptr,
                                 ROWS, DV, DFF, 2);
}
